// CausalSelfAttention_66228395705138
// MI455X (gfx1250) — matrix operand order repaired and checked by reading, not compiled, not run on hardware
//
#include <hip/hip_runtime.h>

#define TLEN 2048
#define CDIM 1024
#define NHEAD 16
#define HDIM 64

#ifndef __has_builtin
#define __has_builtin(x) 0
#endif
#if __has_builtin(__builtin_amdgcn_global_load_async_to_lds_b128) && \
    __has_builtin(__builtin_amdgcn_s_wait_asynccnt)
#define USE_ASYNC 1
#else
#define USE_ASYNC 0
#endif

typedef _Float16 v4h  __attribute__((ext_vector_type(4)));
typedef _Float16 v8h  __attribute__((ext_vector_type(8)));
typedef _Float16 v16h __attribute__((ext_vector_type(16)));
typedef float    v4f  __attribute__((ext_vector_type(4)));
typedef float    v8f  __attribute__((ext_vector_type(8)));

static __device__ __forceinline__ v8f wmma_f16(v16h a, v16h b, v8f c) {
  // D = A(16x32 f16) * B(32x16 f16) + C(16x16 f32)
  return __builtin_amdgcn_wmma_f32_16x16x32_f16(false, a, false, b, (short)0, c,
                                                false, false);
}

// A-fragment: each lane loads 16 contiguous halves:
// K = kbase..kbase+15 (kbase = (lane<16?0:16) + 32*step).
static __device__ __forceinline__ v16h ld_afrag(const _Float16* p) {
  v8h lo = *(const v8h*)p;
  v8h hi = *(const v8h*)(p + 8);
  return __builtin_shufflevector(lo, hi, 0, 1, 2, 3, 4, 5, 6, 7, 8, 9, 10, 11,
                                 12, 13, 14, 15);
}

#if USE_ASYNC
// Parameter types per hipcc diagnostic: (v4i addrspace(1)*, v4i addrspace(3)*,
// imm offset, imm cpol).
typedef int i4 __attribute__((vector_size(4 * sizeof(int))));
typedef __attribute__((address_space(1))) i4* gbl_i4_ptr;
typedef __attribute__((address_space(3))) i4* lds_i4_ptr;

// One lane copies 16 bytes global -> LDS via the CDNA5 async-to-LDS path.
static __device__ __forceinline__ void async_cp16(const void* g, void* l) {
  __builtin_amdgcn_global_load_async_to_lds_b128((gbl_i4_ptr)g, (lds_i4_ptr)l,
                                                 0, 0);
}
#endif

// ---------------- conversion kernels ----------------

__global__ void k_cvt(const float* __restrict__ in, _Float16* __restrict__ out,
                      int n) {
  int i = (blockIdx.x * 256 + threadIdx.x) * 4;
  if (i < n) {
    v4f v = *(const v4f*)(in + i);
    v4h o = {(_Float16)v[0], (_Float16)v[1], (_Float16)v[2], (_Float16)v[3]};
    *(v4h*)(out + i) = o;
  }
}

// W[K][N] f32 -> Wt[N][K] f16 (so GEMM B-fragments read contiguous K)
__global__ void k_cvt_tr(const float* __restrict__ W, _Float16* __restrict__ Wt,
                         int K, int N) {
  int idx = blockIdx.x * 256 + threadIdx.x;
  if (idx < K * N) {
    int k = idx / N, n = idx - k * N;
    Wt[(size_t)n * K + k] = (_Float16)W[idx];
  }
}

// ---------------- shared 128x128 WMMA GEMM core ----------------
// A: [M][K] row-major f16.  Bt: [N][K] row-major f16 (B transposed).
// 256 threads = 8 waves in 4(M) x 2(N); each wave computes 32x64 via 2x4 WMMA
// tiles. Async path: double-buffered LDS with global_load_async_to_lds_b128.
static __device__ __forceinline__ void gemm128x128(
    const _Float16* __restrict__ A, const _Float16* __restrict__ Bt, int K,
    int m0, int n0, v8f acc[2][4]) {
  const int tid = threadIdx.x;
  const int lane = tid & 31, wave = tid >> 5;
  const int lrow = lane & 15, half = lane >> 4;
  const int wm = wave >> 1, wn = wave & 1;
  const int e0 = tid * 8, e1 = e0 + 2048;
  const int r0 = e0 >> 5, c0 = e0 & 31;
  const int r1 = e1 >> 5, c1 = e1 & 31;

#if USE_ASYNC
  __shared__ __align__(32) _Float16 As[2][128 * 32];
  __shared__ __align__(32) _Float16 Bs[2][128 * 32];
  auto stage = [&](int k0, int b) {
    async_cp16(A + (size_t)(m0 + r0) * K + k0 + c0, &As[b][e0]);
    async_cp16(A + (size_t)(m0 + r1) * K + k0 + c1, &As[b][e1]);
    async_cp16(Bt + (size_t)(n0 + r0) * K + k0 + c0, &Bs[b][e0]);
    async_cp16(Bt + (size_t)(n0 + r1) * K + k0 + c1, &Bs[b][e1]);
  };
  int buf = 0;
  stage(0, 0);
  for (int k0 = 0; k0 < K; k0 += 32) {
    __builtin_amdgcn_s_wait_asynccnt(0);  // my copies into `buf` are in LDS
    __syncthreads();                      // everyone's copies are in LDS
    if (k0 + 32 < K) stage(k0 + 32, buf ^ 1);  // prefetch next K-step
    const _Float16* as = &As[buf][0];
    const _Float16* bs = &Bs[buf][0];
    v16h af[2], bf[4];
#pragma unroll
    for (int mt = 0; mt < 2; ++mt)
      af[mt] = ld_afrag(&as[(wm * 32 + mt * 16 + lrow) * 32 + half * 16]);
#pragma unroll
    for (int nt = 0; nt < 4; ++nt)
      bf[nt] = *(const v16h*)&bs[(wn * 64 + nt * 16 + lrow) * 32 + half * 16];
#pragma unroll
    for (int mt = 0; mt < 2; ++mt)
#pragma unroll
      for (int nt = 0; nt < 4; ++nt)
        acc[mt][nt] = wmma_f16(af[mt], bf[nt], acc[mt][nt]);
    buf ^= 1;
  }
#else
  __shared__ __align__(32) _Float16 As[128 * 32];
  __shared__ __align__(32) _Float16 Bs[128 * 32];
  for (int k0 = 0; k0 < K; k0 += 32) {
    __syncthreads();
    *(v8h*)&As[e0] = *(const v8h*)(A + (size_t)(m0 + r0) * K + k0 + c0);
    *(v8h*)&As[e1] = *(const v8h*)(A + (size_t)(m0 + r1) * K + k0 + c1);
    *(v8h*)&Bs[e0] = *(const v8h*)(Bt + (size_t)(n0 + r0) * K + k0 + c0);
    *(v8h*)&Bs[e1] = *(const v8h*)(Bt + (size_t)(n0 + r1) * K + k0 + c1);
    __syncthreads();
    v16h af[2], bf[4];
#pragma unroll
    for (int mt = 0; mt < 2; ++mt)
      af[mt] = ld_afrag(&As[(wm * 32 + mt * 16 + lrow) * 32 + half * 16]);
#pragma unroll
    for (int nt = 0; nt < 4; ++nt)
      bf[nt] = *(const v16h*)&Bs[(wn * 64 + nt * 16 + lrow) * 32 + half * 16];
#pragma unroll
    for (int mt = 0; mt < 2; ++mt)
#pragma unroll
      for (int nt = 0; nt < 4; ++nt)
        acc[mt][nt] = wmma_f16(af[mt], bf[nt], acc[mt][nt]);
  }
#endif
}

// ---------------- QKV projection: X[8192x1024] * Wattn -> Q,K (t-major), V (hd-major)
__global__ void __launch_bounds__(256) k_gemm_qkv(
    const _Float16* __restrict__ Xh, const _Float16* __restrict__ Wt,
    const float* __restrict__ bias, _Float16* __restrict__ Qh,
    _Float16* __restrict__ Kh, _Float16* __restrict__ Vt) {
  const int m0 = blockIdx.y * 128, n0 = blockIdx.x * 128;
  v8f acc[2][4];
#pragma unroll
  for (int mt = 0; mt < 2; ++mt)
#pragma unroll
    for (int nt = 0; nt < 4; ++nt) {
      v8f z = {};
      acc[mt][nt] = z;
    }
  gemm128x128(Xh, Wt, CDIM, m0, n0, acc);

  const int lane = threadIdx.x & 31, wave = threadIdx.x >> 5;
  const int lrow = lane & 15, half = lane >> 4;
  const int wm = wave >> 1, wn = wave & 1;
#pragma unroll
  for (int mt = 0; mt < 2; ++mt)
#pragma unroll
    for (int nt = 0; nt < 4; ++nt) {
      const int jbase = n0 + wn * 64 + nt * 16;  // 16-wide tile: q/k/v-uniform
      const int which = jbase >> 10;             // 0=q, 1=k, 2=v
      const int j = jbase + lrow;
      const float bj = bias[j];
      const int c = j & 1023;
      const int h = c >> 6, d = c & 63;
      _Float16* const base = (which == 0) ? Qh : ((which == 1) ? Kh : Vt);
#pragma unroll
      for (int r = 0; r < 8; ++r) {
        const int i = m0 + wm * 32 + mt * 16 + r + half * 8;  // [0,8192)
        const int b = i >> 11, t = i & 2047;
        const size_t bh = (size_t)b * NHEAD + h;
        const size_t idx_qk = (bh * TLEN + t) * HDIM + d;
        const size_t idx_v = (bh * HDIM + d) * TLEN + t;  // V stored [HD][T]
        base[(which == 2) ? idx_v : idx_qk] = (_Float16)(acc[mt][nt][r] + bj);
      }
    }
}

// ---------------- output projection: Y[8192x1024] * Wproj + b -> f32 out
__global__ void __launch_bounds__(256) k_gemm_proj(
    const _Float16* __restrict__ Yh, const _Float16* __restrict__ Wt,
    const float* __restrict__ bias, float* __restrict__ out) {
  const int m0 = blockIdx.y * 128, n0 = blockIdx.x * 128;
  v8f acc[2][4];
#pragma unroll
  for (int mt = 0; mt < 2; ++mt)
#pragma unroll
    for (int nt = 0; nt < 4; ++nt) {
      v8f z = {};
      acc[mt][nt] = z;
    }
  gemm128x128(Yh, Wt, CDIM, m0, n0, acc);

  const int lane = threadIdx.x & 31, wave = threadIdx.x >> 5;
  const int lrow = lane & 15, half = lane >> 4;
  const int wm = wave >> 1, wn = wave & 1;
#pragma unroll
  for (int mt = 0; mt < 2; ++mt)
#pragma unroll
    for (int nt = 0; nt < 4; ++nt) {
      const int j = n0 + wn * 64 + nt * 16 + lrow;
      const float bj = bias[j];
#pragma unroll
      for (int r = 0; r < 8; ++r) {
        const int i = m0 + wm * 32 + mt * 16 + r + half * 8;
        out[(size_t)i * CDIM + j] = acc[mt][nt][r] + bj;
      }
    }
}

// ---------------- flash attention ----------------
// Grid: B*NH*(T/64) blocks, 128 threads (4 waves). Wave w owns 16 query rows.
// Per 32-key block: S = Q*K^T (4 WMMA), online softmax, O += P*V (4 WMMA).
__global__ void __launch_bounds__(128) k_attn(const _Float16* __restrict__ Qh,
                                              const _Float16* __restrict__ Kh,
                                              const _Float16* __restrict__ Vt,
                                              _Float16* __restrict__ Yh) {
  __shared__ __align__(32) _Float16 Pst[4][16][32];
  const int nqb = TLEN / 64;
  const int bh = blockIdx.x / nqb, qb = blockIdx.x % nqb;
  const int b = bh / NHEAD, h = bh % NHEAD;
  const _Float16* Qb = Qh + (size_t)bh * TLEN * HDIM;
  const _Float16* Kb = Kh + (size_t)bh * TLEN * HDIM;
  const _Float16* Vb = Vt + (size_t)bh * HDIM * TLEN;
  const int wave = threadIdx.x >> 5, lane = threadIdx.x & 31;
  const int lrow = lane & 15, half = lane >> 4;
  const int q0 = qb * 64, qrow0 = q0 + wave * 16;
  const float NEGINF = -__builtin_inff();

  // Q fragments for both 32-wide K-steps over HD=64 (kept in registers)
  v16h qf[2];
  const _Float16* qrow = Qb + (size_t)(qrow0 + lrow) * HDIM;
#pragma unroll
  for (int s = 0; s < 2; ++s) qf[s] = ld_afrag(qrow + s * 32 + half * 16);

  v8f o[4];
  float mrun[8], lrun[8];
#pragma unroll
  for (int ht = 0; ht < 4; ++ht) {
    v8f z = {};
    o[ht] = z;
  }
#pragma unroll
  for (int r = 0; r < 8; ++r) {
    mrun[r] = NEGINF;
    lrun[r] = 0.0f;
  }

  for (int kb = 0; kb < q0 + 64; kb += 32) {
    const bool active = (kb < qrow0 + 16);  // wave-uniform
    if (active) {
      v8f s0 = {}, s1 = {};
#pragma unroll
      for (int s = 0; s < 2; ++s) {
        // B-fragment of K^T: lane = key column, contiguous 16 hd values
        v16h b0 =
            *(const v16h*)(Kb + (size_t)(kb + lrow) * HDIM + s * 32 + half * 16);
        v16h b1 = *(const v16h*)(Kb + (size_t)(kb + 16 + lrow) * HDIM + s * 32 +
                                 half * 16);
        s0 = wmma_f16(qf[s], b0, s0);
        s1 = wmma_f16(qf[s], b1, s1);
      }
      const float sc = 0.125f;  // 1/sqrt(64)
#pragma unroll
      for (int r = 0; r < 8; ++r) {
        int row = qrow0 + r + half * 8;
        float v0 = (kb + lrow <= row) ? s0[r] * sc : NEGINF;
        float v1 = (kb + 16 + lrow <= row) ? s1[r] * sc : NEGINF;
        float bm = fmaxf(v0, v1);
        bm = fmaxf(bm, __shfl_xor(bm, 1));
        bm = fmaxf(bm, __shfl_xor(bm, 2));
        bm = fmaxf(bm, __shfl_xor(bm, 4));
        bm = fmaxf(bm, __shfl_xor(bm, 8));
        float nm = fmaxf(mrun[r], bm);
        float p0 = __expf(v0 - nm);
        float p1 = __expf(v1 - nm);
        float rs = p0 + p1;
        rs += __shfl_xor(rs, 1);
        rs += __shfl_xor(rs, 2);
        rs += __shfl_xor(rs, 4);
        rs += __shfl_xor(rs, 8);
        float scale = __expf(mrun[r] - nm);
        lrun[r] = lrun[r] * scale + rs;
        mrun[r] = nm;
#pragma unroll
        for (int ht = 0; ht < 4; ++ht) o[ht][r] *= scale;
        Pst[wave][r + half * 8][lrow] = (_Float16)p0;
        Pst[wave][r + half * 8][lrow + 16] = (_Float16)p1;
      }
    }
    __syncthreads();
    if (active) {
      v16h pa = ld_afrag(&Pst[wave][lrow][half * 16]);
#pragma unroll
      for (int ht = 0; ht < 4; ++ht) {
        // B-fragment of V: Vt[hd][t] rows are key-contiguous
        v16h vb = *(const v16h*)(Vb + (size_t)(ht * 16 + lrow) * TLEN + kb +
                                 half * 16);
        o[ht] = wmma_f16(pa, vb, o[ht]);
      }
    }
    __syncthreads();
  }

#pragma unroll
  for (int ht = 0; ht < 4; ++ht)
#pragma unroll
    for (int r = 0; r < 8; ++r) {
      int row = qrow0 + r + half * 8;
      float v = o[ht][r] / lrun[r];
      Yh[((size_t)b * TLEN + row) * CDIM + h * HDIM + ht * 16 + lrow] =
          (_Float16)v;
    }
}

// ---------------- launch ----------------
extern "C" void kernel_launch(void* const* d_in, const int* in_sizes, int n_in,
                              void* d_out, int out_size, void* d_ws,
                              size_t ws_size, hipStream_t stream) {
  (void)in_sizes;
  (void)n_in;
  (void)out_size;
  (void)ws_size;
  const float* x = (const float*)d_in[0];
  const float* W_attn = (const float*)d_in[1];
  const float* b_attn = (const float*)d_in[2];
  const float* W_proj = (const float*)d_in[3];
  const float* b_proj = (const float*)d_in[4];
  float* out = (float*)d_out;

  char* ws = (char*)d_ws;
  size_t off = 0;
  auto alloc = [&](size_t bytes) -> void* {
    void* p = ws + off;
    off += (bytes + 255) & ~(size_t)255;
    return p;
  };
  _Float16* Xh = (_Float16*)alloc((size_t)8192 * 1024 * 2);   // x in f16
  _Float16* Wah = (_Float16*)alloc((size_t)3072 * 1024 * 2);  // W_attn^T f16
  _Float16* Wph = (_Float16*)alloc((size_t)1024 * 1024 * 2);  // W_proj^T f16
  _Float16* Qh = (_Float16*)alloc((size_t)64 * 2048 * 64 * 2);
  _Float16* Kh = (_Float16*)alloc((size_t)64 * 2048 * 64 * 2);
  _Float16* Vt = (_Float16*)alloc((size_t)64 * 2048 * 64 * 2);
  _Float16* Yh = Xh;  // reuse: x no longer needed after the QKV GEMM

  k_cvt<<<8192 * 1024 / 4 / 256, 256, 0, stream>>>(x, Xh, 8192 * 1024);
  k_cvt_tr<<<(1024 * 3072 + 255) / 256, 256, 0, stream>>>(W_attn, Wah, 1024,
                                                          3072);
  k_cvt_tr<<<(1024 * 1024 + 255) / 256, 256, 0, stream>>>(W_proj, Wph, 1024,
                                                          1024);
  k_gemm_qkv<<<dim3(3072 / 128, 8192 / 128), 256, 0, stream>>>(Xh, Wah, b_attn,
                                                               Qh, Kh, Vt);
  k_attn<<<dim3(4 * NHEAD * (TLEN / 64)), 128, 0, stream>>>(Qh, Kh, Vt, Yh);
  k_gemm_proj<<<dim3(1024 / 128, 8192 / 128), 256, 0, stream>>>(Yh, Wph, b_proj,
                                                                out);
}
